// TemplatePairStack_2963527434516
// MI455X (gfx1250) — matrix operand order repaired and checked by reading, not compiled, not run on hardware
//
#include <hip/hip_runtime.h>
#include <hip/hip_bf16.h>

// ---------------------------------------------------------------------------
// AF2 TemplatePairStack for gfx1250 (CDNA5 / MI455X).
// All contractions run on v_wmma_f32_16x16x32_f16 (f16 in, f32 accumulate),
// converting f32 operands to f16 at fragment-load time from LDS.
//
// Input flattening assumption (JAX pytree, sorted dict keys inside params):
//   d_in[0] = t [T,N,N,C] f32
//   d_in[1] = mask [T,N,N] f32
//   d_in[2..11]  att_end : b_w,g_b,g_w,k_w,ln_b,ln_g,o_b,o_w,q_w,v_w
//   d_in[12..21] att_start (same order)
//   d_in[22] final_ln_b, d_in[23] final_ln_g
//   d_in[24..39] mul_in : ag_b,ag_w,ap_b,ap_w,bg_b,bg_w,bp_b,bp_w,
//                         g_b,g_w,ln_in_b,ln_in_g,ln_out_b,ln_out_g,o_b,o_w
//   d_in[40..55] mul_out (same order)
//   d_in[56..61] trans : b1,b2,ln_b,ln_g,w1,w2
// ---------------------------------------------------------------------------

#define LBLK 2
#define TT   4
#define NN   192
#define CC   64
#define CHH  64
#define HH   4
#define CKK  16
#define CTT  128
#define N2   (NN * NN)
#define RR   (TT * N2)          // 147456 rows
#define RC   ((size_t)RR * 64)  // elements of a [T,N,N,64] tensor

typedef __attribute__((ext_vector_type(16))) _Float16 v16h;
typedef __attribute__((ext_vector_type(8)))  float    v8f;

static __device__ __forceinline__ float sigmoidf_(float x) {
  return 1.0f / (1.0f + __expf(-x));
}

// ---------------- WMMA fragment loaders (contiguous k = 16*hi + s, same for A and B) ----------------
// A-matrix 16x32 f16: lane m=lane&15, hi=lane>>4;
//   v : K = 16*hi + 2v, 2v+1
static __device__ __forceinline__ v16h frag_a_f32(const float* src, int ld, int lane) {
  int m = lane & 15, hi = lane >> 4;
  v16h a;
#pragma unroll
  for (int v = 0; v < 8; ++v) {
    int kb = 16 * hi + 2 * v;
    const float* p = src + m * ld + kb;
    a[2 * v]     = (_Float16)p[0];
    a[2 * v + 1] = (_Float16)p[1];
  }
  return a;
}
// B-matrix 32x16 f16 (K-major [K][N] tile): lane n=lane&15, hi=lane>>4;
//   v : K = 16*hi + 2v, 2v+1
static __device__ __forceinline__ v16h frag_b_f32(const float* src, int ld, int lane) {
  int n = lane & 15, hi = lane >> 4;
  v16h b;
#pragma unroll
  for (int v = 0; v < 8; ++v) {
    int kk = 16 * hi + 2 * v;
    b[2 * v]     = (_Float16)src[kk * ld + n];
    b[2 * v + 1] = (_Float16)src[(kk + 1) * ld + n];
  }
  return b;
}
// B fragment from an N-major [N][K] tile (i.e. operand is stored transposed)
static __device__ __forceinline__ v16h frag_bt_f32(const float* src, int ld, int lane) {
  int n = lane & 15, hi = lane >> 4;
  v16h b;
#pragma unroll
  for (int v = 0; v < 8; ++v) {
    const float* p = src + n * ld + 16 * hi + 2 * v;
    b[2 * v]     = (_Float16)p[0];
    b[2 * v + 1] = (_Float16)p[1];
  }
  return b;
}

// --------------------------- batched tiled GEMM -----------------------------
// C[M,Nn] (+)= A[M,K] * B[K,Nn]   (f32 storage, f16 WMMA math, f32 accum)
// EPI: 0 store, 1 relu-store, 2 add, 3 sigmoid(gate)*(..) add, 4 rowscale add
#define AS_LD 34
#define BS_LD 66

template <int EPI, bool TB>
__global__ __launch_bounds__(256) void gemm_kernel(
    const float* __restrict__ A, long long sAb, int lda,
    const float* __restrict__ B, long long sBb, int ldb,
    float* __restrict__ C, long long sCb, int ldc,
    const float* __restrict__ bias,
    const float* __restrict__ gate,
    const float* __restrict__ rowscale,
    int M, int Nc, int K) {
  __shared__ float As[64 * AS_LD];
  __shared__ float Bs[32 * BS_LD];

  const int tid = threadIdx.x, wave = tid >> 5, lane = tid & 31;
  const int m0 = blockIdx.x * 64, n0 = blockIdx.y * 64;
  const long long zb = blockIdx.z;
  const float* Ab = A + zb * sAb;
  const float* Bb = B + zb * sBb;
  float* Cb = C + zb * sCb;

  const int tile_r = wave & 3;       // 0..3 : 16-row tile within 64
  const int colgrp = wave >> 2;      // 0..1 : covers cols {g*16, g*16+32}
  v8f acc0 = {0.f, 0.f, 0.f, 0.f, 0.f, 0.f, 0.f, 0.f};
  v8f acc1 = {0.f, 0.f, 0.f, 0.f, 0.f, 0.f, 0.f, 0.f};

  for (int k0 = 0; k0 < K; k0 += 32) {
    // ---- stage A tile [64 x 32] ----
    {
      int r = tid >> 2, kq = (tid & 3) * 8;
      int gm = m0 + r;
      float* dst = &As[r * AS_LD + kq];
      if (gm < M && (k0 + kq + 8) <= K && ((lda & 3) == 0)) {
        const float4* p = reinterpret_cast<const float4*>(Ab + (size_t)gm * lda + k0 + kq);
        float4 x0 = p[0], x1 = p[1];
        dst[0] = x0.x; dst[1] = x0.y; dst[2] = x0.z; dst[3] = x0.w;
        dst[4] = x1.x; dst[5] = x1.y; dst[6] = x1.z; dst[7] = x1.w;
      } else {
#pragma unroll
        for (int u = 0; u < 8; ++u) {
          int kk = k0 + kq + u;
          dst[u] = (gm < M && kk < K) ? Ab[(size_t)gm * lda + kk] : 0.f;
        }
      }
    }
    // ---- stage B tile [32 x 64] ----
    if (!TB) {
      int kk = tid >> 3, nq = (tid & 7) * 8;
      int gk = k0 + kk;
      float* dst = &Bs[kk * BS_LD + nq];
      if (gk < K && (n0 + nq + 8) <= Nc && ((ldb & 3) == 0)) {
        const float4* p = reinterpret_cast<const float4*>(Bb + (size_t)gk * ldb + n0 + nq);
        float4 x0 = p[0], x1 = p[1];
        dst[0] = x0.x; dst[1] = x0.y; dst[2] = x0.z; dst[3] = x0.w;
        dst[4] = x1.x; dst[5] = x1.y; dst[6] = x1.z; dst[7] = x1.w;
      } else {
#pragma unroll
        for (int u = 0; u < 8; ++u) {
          int gn = n0 + nq + u;
          dst[u] = (gk < K && gn < Nc) ? Bb[(size_t)gk * ldb + gn] : 0.f;
        }
      }
    } else {
      // B stored transposed: [Nn][K]; Bs[k][n] = Bst[n0+n][k0+k]
      int n = tid >> 2, kq = (tid & 3) * 8;
      int gn = n0 + n;
      if (gn < Nc && (k0 + kq + 8) <= K && ((ldb & 3) == 0)) {
        const float4* p = reinterpret_cast<const float4*>(Bb + (size_t)gn * ldb + k0 + kq);
        float4 x0 = p[0], x1 = p[1];
        float tmp[8] = {x0.x, x0.y, x0.z, x0.w, x1.x, x1.y, x1.z, x1.w};
#pragma unroll
        for (int u = 0; u < 8; ++u) Bs[(kq + u) * BS_LD + n] = tmp[u];
      } else {
#pragma unroll
        for (int u = 0; u < 8; ++u) {
          int kk = k0 + kq + u;
          Bs[(kq + u) * BS_LD + n] = (gn < Nc && kk < K) ? Bb[(size_t)gn * ldb + kk] : 0.f;
        }
      }
    }
    __syncthreads();

    v16h af  = frag_a_f32(&As[tile_r * 16 * AS_LD], AS_LD, lane);
    v16h bf0 = frag_b_f32(&Bs[colgrp * 16], BS_LD, lane);
    v16h bf1 = frag_b_f32(&Bs[colgrp * 16 + 32], BS_LD, lane);
    acc0 = __builtin_amdgcn_wmma_f32_16x16x32_f16(false, af, false, bf0, (short)0, acc0, false, false);
    acc1 = __builtin_amdgcn_wmma_f32_16x16x32_f16(false, af, false, bf1, (short)0, acc1, false, false);
    __syncthreads();
  }

  // ---- epilogue: C/D layout (lanes 0-15: M=r, lanes 16-31: M=8+r) ----
  const int hi = lane >> 4, nloc = lane & 15;
  const int mrow_base = m0 + tile_r * 16 + hi * 8;
#pragma unroll
  for (int tle = 0; tle < 2; ++tle) {
    int col = n0 + colgrp * 16 + tle * 32 + nloc;
    v8f& acc = tle ? acc1 : acc0;
    if (col < Nc) {
#pragma unroll
      for (int r = 0; r < 8; ++r) {
        int row = mrow_base + r;
        if (row < M) {
          float v = acc[r];
          if (bias) v += bias[col];
          size_t off = (size_t)row * ldc + col;
          if (EPI == 0)      Cb[off] = v;
          else if (EPI == 1) Cb[off] = fmaxf(v, 0.f);
          else if (EPI == 2) Cb[off] += v;
          else if (EPI == 3) Cb[off] += sigmoidf_(gate[off]) * v;
          else if (EPI == 4) Cb[off] += rowscale[row] * v;
        }
      }
    }
  }
}

static void launch_gemm(hipStream_t s, int epi, bool tb,
                        const float* A, long long sAb, int lda,
                        const float* B, long long sBb, int ldb,
                        float* C, long long sCb, int ldc,
                        const float* bias, const float* gate, const float* rowscale,
                        int M, int Nc, int K, int batch) {
  dim3 g((M + 63) / 64, (Nc + 63) / 64, batch);
  if (tb) {
    gemm_kernel<0, true><<<g, 256, 0, s>>>(A, sAb, lda, B, sBb, ldb, C, sCb, ldc, bias, gate, rowscale, M, Nc, K);
    return;
  }
  switch (epi) {
    case 0: gemm_kernel<0, false><<<g, 256, 0, s>>>(A, sAb, lda, B, sBb, ldb, C, sCb, ldc, bias, gate, rowscale, M, Nc, K); break;
    case 1: gemm_kernel<1, false><<<g, 256, 0, s>>>(A, sAb, lda, B, sBb, ldb, C, sCb, ldc, bias, gate, rowscale, M, Nc, K); break;
    case 2: gemm_kernel<2, false><<<g, 256, 0, s>>>(A, sAb, lda, B, sBb, ldb, C, sCb, ldc, bias, gate, rowscale, M, Nc, K); break;
    case 3: gemm_kernel<3, false><<<g, 256, 0, s>>>(A, sAb, lda, B, sBb, ldb, C, sCb, ldc, bias, gate, rowscale, M, Nc, K); break;
    default: gemm_kernel<4, false><<<g, 256, 0, s>>>(A, sAb, lda, B, sBb, ldb, C, sCb, ldc, bias, gate, rowscale, M, Nc, K); break;
  }
}

// ------------------------------ LayerNorm (C=64) ----------------------------
__global__ __launch_bounds__(256) void ln_kernel(const float* __restrict__ x,
                                                 const float* __restrict__ g,
                                                 const float* __restrict__ b,
                                                 float* __restrict__ y, int rows) {
  int wave = threadIdx.x >> 5, lane = threadIdx.x & 31;
  int row = blockIdx.x * 8 + wave;
  if (row >= rows) return;
  const float* xr = x + (size_t)row * 64;
  float v0 = xr[lane], v1 = xr[lane + 32];
  float sm = v0 + v1;
#pragma unroll
  for (int o = 16; o; o >>= 1) sm += __shfl_xor(sm, o, 32);
  float mu = sm * (1.0f / 64.0f);
  float d0 = v0 - mu, d1 = v1 - mu;
  float q = d0 * d0 + d1 * d1;
#pragma unroll
  for (int o = 16; o; o >>= 1) q += __shfl_xor(q, o, 32);
  float rs = rsqrtf(q * (1.0f / 64.0f) + 1e-5f);
  float* yr = y + (size_t)row * 64;
  yr[lane]      = d0 * rs * g[lane]      + b[lane];
  yr[lane + 32] = d1 * rs * g[lane + 32] + b[lane + 32];
}

// --- combine: dst[t][c][i][k] = sigmoid(G+gb)*(P+pb)*mask (bias fused in GEMM)
__global__ __launch_bounds__(256) void combine_ab_kernel(const float* __restrict__ G,
                                                         const float* __restrict__ P,
                                                         const float* __restrict__ mask,
                                                         float* __restrict__ dst, int swap) {
  size_t idx = (size_t)blockIdx.x * 256 + threadIdx.x;  // over T*CH*N*N
  int t = idx / ((size_t)CHH * N2);
  size_t rem = idx % ((size_t)CHH * N2);
  int c = rem / N2;
  int r2 = rem % N2;
  int i = r2 / NN, k = r2 % NN;
  int si = swap ? k : i;
  int sj = swap ? i : k;
  size_t sr = ((size_t)t * NN + si) * NN + sj;
  dst[idx] = sigmoidf_(G[sr * 64 + c]) * P[sr * 64 + c] * mask[sr];
}

// --- LayerNorm over channel-major einsum output: X[T,CH,N,N] -> Y[T,N,N,CH]
__global__ __launch_bounds__(256) void xln_kernel(const float* __restrict__ X,
                                                  const float* __restrict__ g,
                                                  const float* __restrict__ b,
                                                  float* __restrict__ Y) {
  size_t gid = (size_t)blockIdx.x * 256 + threadIdx.x;  // over T*N*N
  int t = gid / N2;
  int r2 = gid % N2;
  size_t base = (size_t)t * 64 * N2 + r2;
  float v[64];
  float sm = 0.f;
#pragma unroll
  for (int c = 0; c < 64; ++c) { v[c] = X[base + (size_t)c * N2]; sm += v[c]; }
  float mu = sm * (1.0f / 64.0f);
  float q = 0.f;
#pragma unroll
  for (int c = 0; c < 64; ++c) { float d = v[c] - mu; q += d * d; }
  float rs = rsqrtf(q * (1.0f / 64.0f) + 1e-5f);
  float* yr = Y + gid * 64;
#pragma unroll
  for (int c = 0; c < 64; ++c) yr[c] = (v[c] - mu) * rs * g[c] + b[c];
}

__global__ __launch_bounds__(256) void gate_mul_kernel(const float* __restrict__ Gz,
                                                       const float* __restrict__ O,
                                                       float* __restrict__ GO) {
  size_t i = (size_t)blockIdx.x * 256 + threadIdx.x;
  GO[i] = sigmoidf_(Gz[i]) * O[i];
}

// --- transpose [T,N,N,64] over (i,j), float4 over channels
__global__ __launch_bounds__(256) void transpose_kernel(const float* __restrict__ src,
                                                        float* __restrict__ dst) {
  size_t idx = (size_t)blockIdx.x * 256 + threadIdx.x;  // over T*N*N*16 float4
  int t = idx / ((size_t)N2 * 16);
  size_t rem = idx % ((size_t)N2 * 16);
  int i = rem / (NN * 16);
  size_t rem2 = rem % (NN * 16);
  int j = rem2 / 16, c4 = rem2 % 16;
  const float4* s4 = reinterpret_cast<const float4*>(src);
  float4* d4 = reinterpret_cast<float4*>(dst);
  d4[(((size_t)t * NN + j) * NN + i) * 16 + c4] = s4[idx];
}

__global__ __launch_bounds__(256) void mask_t_kernel(const float* __restrict__ src,
                                                     float* __restrict__ dst) {
  size_t idx = (size_t)blockIdx.x * 256 + threadIdx.x;  // over T*N*N
  int t = idx / N2;
  int i = (idx % N2) / NN, j = idx % NN;
  dst[((size_t)t * NN + j) * NN + i] = src[idx];
}

// ------------------- fused triangle attention (per t,h,i,j-tile) ------------
// Q/K/V: [T,N,N,64] f32 (head h at cols h*16..h*16+15); Pb: [T,N,N,4]; O same as Q.
__global__ __launch_bounds__(256) void attn_kernel(const float* __restrict__ Q,
                                                   const float* __restrict__ Km,
                                                   const float* __restrict__ Vm,
                                                   const float* __restrict__ Pb,
                                                   const float* __restrict__ mask,
                                                   float* __restrict__ O) {
  __shared__ float q_s[16 * 32];
  __shared__ float k_s[192 * 32];
  __shared__ float v_s[192 * 17];
  __shared__ float Ssm[16 * 196];
  __shared__ float Op[6 * 16 * 17];

  const int jt = blockIdx.x;                 // 0..11
  const int i  = blockIdx.y;                 // 0..191
  const int t  = blockIdx.z / HH;
  const int h  = blockIdx.z % HH;
  const int tid = threadIdx.x, wave = tid >> 5, lane = tid & 31;
  const size_t base = ((size_t)t * NN + i) * NN * 64;

  for (int idx = tid; idx < 16 * 32; idx += 256) {
    int r = idx >> 5, c = idx & 31;
    q_s[idx] = (c < 16) ? Q[base + (size_t)(jt * 16 + r) * 64 + h * 16 + c] : 0.f;
  }
  for (int idx = tid; idx < 192 * 32; idx += 256) {
    int r = idx >> 5, c = idx & 31;
    k_s[idx] = (c < 16) ? Km[base + (size_t)r * 64 + h * 16 + c] : 0.f;
  }
  for (int idx = tid; idx < 192 * 16; idx += 256) {
    int r = idx >> 4, c = idx & 15;
    v_s[r * 17 + c] = Vm[base + (size_t)r * 64 + h * 16 + c];
  }
  __syncthreads();

  const int hi = lane >> 4, nloc = lane & 15;
  // phase 1: S = q k^T / 4 + bias + mask
  for (int kt = wave; kt < 12; kt += 8) {
    v16h a  = frag_a_f32(q_s, 32, lane);
    v16h bf = frag_bt_f32(&k_s[(kt * 16) * 32], 32, lane);
    v8f acc = {0.f, 0.f, 0.f, 0.f, 0.f, 0.f, 0.f, 0.f};
    acc = __builtin_amdgcn_wmma_f32_16x16x32_f16(false, a, false, bf, (short)0, acc, false, false);
    int kg = kt * 16 + nloc;
    float mk = (mask[((size_t)t * NN + i) * NN + kg] - 1.0f) * 1e9f;
#pragma unroll
    for (int r = 0; r < 8; ++r) {
      int mrow = hi * 8 + r;
      float pb = Pb[(((size_t)t * NN + (jt * 16 + mrow)) * NN + kg) * HH + h];
      Ssm[mrow * 196 + kg] = acc[r] * 0.25f + pb + mk;
    }
  }
  __syncthreads();

  // softmax over k (192 cols) per row; wave handles rows 2w, 2w+1
#pragma unroll
  for (int rr = 0; rr < 2; ++rr) {
    int mrow = wave * 2 + rr;
    float vals[6];
    float mx = -1e30f;
#pragma unroll
    for (int q = 0; q < 6; ++q) { vals[q] = Ssm[mrow * 196 + lane + q * 32]; mx = fmaxf(mx, vals[q]); }
#pragma unroll
    for (int o = 16; o; o >>= 1) mx = fmaxf(mx, __shfl_xor(mx, o, 32));
    float sm = 0.f;
#pragma unroll
    for (int q = 0; q < 6; ++q) { vals[q] = __expf(vals[q] - mx); sm += vals[q]; }
#pragma unroll
    for (int o = 16; o; o >>= 1) sm += __shfl_xor(sm, o, 32);
    float inv = 1.0f / sm;
#pragma unroll
    for (int q = 0; q < 6; ++q) Ssm[mrow * 196 + lane + q * 32] = vals[q] * inv;
  }
  __syncthreads();

  // phase 2: O_tile = attn @ V ; 6 K-chunks of 32 across waves 0..5
  if (wave < 6) {
    v16h a = frag_a_f32(&Ssm[wave * 32], 196, lane);
    v16h b = frag_b_f32(&v_s[wave * 32 * 17], 17, lane);
    v8f acc = {0.f, 0.f, 0.f, 0.f, 0.f, 0.f, 0.f, 0.f};
    acc = __builtin_amdgcn_wmma_f32_16x16x32_f16(false, a, false, b, (short)0, acc, false, false);
#pragma unroll
    for (int r = 0; r < 8; ++r) Op[(wave * 16 + hi * 8 + r) * 17 + nloc] = acc[r];
  }
  __syncthreads();

  int m = tid >> 4, c = tid & 15;
  float s = 0.f;
#pragma unroll
  for (int p = 0; p < 6; ++p) s += Op[(p * 16 + m) * 17 + c];
  O[base + (size_t)(jt * 16 + m) * 64 + h * 16 + c] = s;
}

// ------------------------------- host side ----------------------------------
// param index map (see header comment)
enum { ATT_END = 2, ATT_START = 12, FINAL_LN_B = 22, FINAL_LN_G = 23,
       MUL_IN = 24, MUL_OUT = 40, TRANSB = 56 };
enum { A_B_W = 0, A_G_B, A_G_W, A_K_W, A_LN_B, A_LN_G, A_O_B, A_O_W, A_Q_W, A_V_W };
enum { M_AG_B = 0, M_AG_W, M_AP_B, M_AP_W, M_BG_B, M_BG_W, M_BP_B, M_BP_W,
       M_G_B, M_G_W, M_LN_IN_B, M_LN_IN_G, M_LN_OUT_B, M_LN_OUT_G, M_O_B, M_O_W };
enum { TR_B1 = 0, TR_B2, TR_LN_B, TR_LN_G, TR_W1, TR_W2 };

static const int ATT_SZ[10] = {256, 64, 4096, 4096, 64, 64, 64, 4096, 4096, 4096};
static const int MUL_SZ[16] = {64, 4096, 64, 4096, 64, 4096, 64, 4096, 64, 4096, 64, 64, 64, 64, 64, 4096};
static const int TRN_SZ[6]  = {128, 64, 64, 64, 8192, 8192};

struct Ws {
  float *Tbuf, *Tt, *Z, *A1, *A2, *A3, *A4, *A5, *A6, *Pb, *maskT;
};

static inline const float* PRM(void* const* d_in, int base, int rel, int l, const int* szs) {
  return ((const float*)d_in[base + rel]) + (size_t)l * szs[rel];
}

static void run_trimul(hipStream_t s, void* const* d_in, int base, int l,
                       float* buf, const float* mask, Ws& w, bool outgoing) {
  const int* SZ = MUL_SZ;
  // z = LN(buf)
  ln_kernel<<<dim3(RR / 8), 256, 0, s>>>(buf, PRM(d_in, base, M_LN_IN_G, l, SZ),
                                         PRM(d_in, base, M_LN_IN_B, l, SZ), w.Z, RR);
  // a projections
  launch_gemm(s, 0, false, w.Z, 0, 64, PRM(d_in, base, M_AG_W, l, SZ), 0, 64,
              w.A1, 0, 64, PRM(d_in, base, M_AG_B, l, SZ), nullptr, nullptr, RR, 64, 64, 1);
  launch_gemm(s, 0, false, w.Z, 0, 64, PRM(d_in, base, M_AP_W, l, SZ), 0, 64,
              w.A2, 0, 64, PRM(d_in, base, M_AP_B, l, SZ), nullptr, nullptr, RR, 64, 64, 1);
  combine_ab_kernel<<<dim3((unsigned)(RC / 256)), 256, 0, s>>>(w.A1, w.A2, mask, w.A4, outgoing ? 0 : 1);
  // b projections
  launch_gemm(s, 0, false, w.Z, 0, 64, PRM(d_in, base, M_BG_W, l, SZ), 0, 64,
              w.A1, 0, 64, PRM(d_in, base, M_BG_B, l, SZ), nullptr, nullptr, RR, 64, 64, 1);
  launch_gemm(s, 0, false, w.Z, 0, 64, PRM(d_in, base, M_BP_W, l, SZ), 0, 64,
              w.A2, 0, 64, PRM(d_in, base, M_BP_B, l, SZ), nullptr, nullptr, RR, 64, 64, 1);
  combine_ab_kernel<<<dim3((unsigned)(RC / 256)), 256, 0, s>>>(w.A1, w.A2, mask, w.A5, outgoing ? 0 : 1);
  // gate raw (z @ g_w + g_b)
  launch_gemm(s, 0, false, w.Z, 0, 64, PRM(d_in, base, M_G_W, l, SZ), 0, 64,
              w.A3, 0, 64, PRM(d_in, base, M_G_B, l, SZ), nullptr, nullptr, RR, 64, 64, 1);
  // einsum: X[t,c][i,j] = sum_k A4[t,c][i,k] * A5[t,c][j,k]  (batched, trans-B)
  launch_gemm(s, 0, true, w.A4, (long long)N2, NN, w.A5, (long long)N2, NN,
              w.A6, (long long)N2, NN, nullptr, nullptr, nullptr, NN, NN, NN, TT * CHH);
  // LN over c, back to [T,N,N,CH]
  xln_kernel<<<dim3(RR / 256), 256, 0, s>>>(w.A6, PRM(d_in, base, M_LN_OUT_G, l, SZ),
                                            PRM(d_in, base, M_LN_OUT_B, l, SZ), w.A1);
  // buf += sigmoid(gate) * (x_ln @ o_w + o_b)
  launch_gemm(s, 3, false, w.A1, 0, 64, PRM(d_in, base, M_O_W, l, SZ), 0, 64,
              buf, 0, 64, PRM(d_in, base, M_O_B, l, SZ), w.A3, nullptr, RR, 64, 64, 1);
}

static void run_triatt(hipStream_t s, void* const* d_in, int base, int l,
                       float* buf, const float* mask, Ws& w) {
  const int* SZ = ATT_SZ;
  ln_kernel<<<dim3(RR / 8), 256, 0, s>>>(buf, PRM(d_in, base, A_LN_G, l, SZ),
                                         PRM(d_in, base, A_LN_B, l, SZ), w.Z, RR);
  launch_gemm(s, 0, false, w.Z, 0, 64, PRM(d_in, base, A_Q_W, l, SZ), 0, 64,
              w.A1, 0, 64, nullptr, nullptr, nullptr, RR, 64, 64, 1);
  launch_gemm(s, 0, false, w.Z, 0, 64, PRM(d_in, base, A_K_W, l, SZ), 0, 64,
              w.A2, 0, 64, nullptr, nullptr, nullptr, RR, 64, 64, 1);
  launch_gemm(s, 0, false, w.Z, 0, 64, PRM(d_in, base, A_V_W, l, SZ), 0, 64,
              w.A3, 0, 64, nullptr, nullptr, nullptr, RR, 64, 64, 1);
  launch_gemm(s, 0, false, w.Z, 0, 64, PRM(d_in, base, A_B_W, l, SZ), 0, HH,
              w.Pb, 0, HH, nullptr, nullptr, nullptr, RR, HH, 64, 1);
  launch_gemm(s, 0, false, w.Z, 0, 64, PRM(d_in, base, A_G_W, l, SZ), 0, 64,
              w.A4, 0, 64, PRM(d_in, base, A_G_B, l, SZ), nullptr, nullptr, RR, 64, 64, 1);
  attn_kernel<<<dim3(12, NN, TT * HH), 256, 0, s>>>(w.A1, w.A2, w.A3, w.Pb, mask, w.A5);
  gate_mul_kernel<<<dim3((unsigned)(RC / 256)), 256, 0, s>>>(w.A4, w.A5, w.A6);
  launch_gemm(s, 2, false, w.A6, 0, 64, PRM(d_in, base, A_O_W, l, SZ), 0, 64,
              buf, 0, 64, PRM(d_in, base, A_O_B, l, SZ), nullptr, nullptr, RR, 64, 64, 1);
}

static void run_transition(hipStream_t s, void* const* d_in, int l,
                           float* buf, const float* mask, Ws& w) {
  const int* SZ = TRN_SZ;
  ln_kernel<<<dim3(RR / 8), 256, 0, s>>>(buf, PRM(d_in, TRANSB, TR_LN_G, l, SZ),
                                         PRM(d_in, TRANSB, TR_LN_B, l, SZ), w.Z, RR);
  // h = relu(z @ w1 + b1) : [R x 128] spans A1+A2 (contiguous)
  launch_gemm(s, 1, false, w.Z, 0, 64, PRM(d_in, TRANSB, TR_W1, l, SZ), 0, CTT,
              w.A1, 0, CTT, PRM(d_in, TRANSB, TR_B1, l, SZ), nullptr, nullptr, RR, CTT, 64, 1);
  // buf += mask_row * (h @ w2 + b2)
  launch_gemm(s, 4, false, w.A1, 0, CTT, PRM(d_in, TRANSB, TR_W2, l, SZ), 0, 64,
              buf, 0, 64, PRM(d_in, TRANSB, TR_B2, l, SZ), nullptr, mask, RR, 64, CTT, 1);
}

extern "C" void kernel_launch(void* const* d_in, const int* in_sizes, int n_in,
                              void* d_out, int out_size, void* d_ws, size_t ws_size,
                              hipStream_t stream) {
  (void)in_sizes; (void)n_in; (void)out_size; (void)ws_size;
  const float* t_in  = (const float*)d_in[0];
  const float* mask  = (const float*)d_in[1];
  float* out = (float*)d_out;

  // workspace carve (floats)
  float* p = (float*)d_ws;
  Ws w;
  w.Tbuf = p; p += RC;
  w.Tt   = p; p += RC;
  w.Z    = p; p += RC;
  w.A1   = p; p += RC;
  w.A2   = p; p += RC;   // A1+A2 contiguous -> transition hidden [R x 128]
  w.A3   = p; p += RC;
  w.A4   = p; p += RC;
  w.A5   = p; p += RC;
  w.A6   = p; p += RC;
  w.Pb   = p; p += (size_t)RR * HH;
  w.maskT = p; p += (size_t)RR;

  hipMemcpyAsync(w.Tbuf, t_in, RC * sizeof(float), hipMemcpyDeviceToDevice, stream);
  mask_t_kernel<<<dim3(RR / 256), 256, 0, stream>>>(mask, w.maskT);

  for (int l = 0; l < LBLK; ++l) {
    run_trimul(stream, d_in, MUL_OUT, l, w.Tbuf, mask, w, true);
    run_trimul(stream, d_in, MUL_IN,  l, w.Tbuf, mask, w, false);
    run_triatt(stream, d_in, ATT_START, l, w.Tbuf, mask, w);
    transpose_kernel<<<dim3((unsigned)(RR * 16 / 256)), 256, 0, stream>>>(w.Tbuf, w.Tt);
    run_triatt(stream, d_in, ATT_END, l, w.Tt, w.maskT, w);
    transpose_kernel<<<dim3((unsigned)(RR * 16 / 256)), 256, 0, stream>>>(w.Tt, w.Tbuf);
    run_transition(stream, d_in, l, w.Tbuf, mask, w);
  }

  ln_kernel<<<dim3(RR / 8), 256, 0, stream>>>(w.Tbuf, (const float*)d_in[FINAL_LN_G],
                                              (const float*)d_in[FINAL_LN_B], out, RR);
}
